// SoftMoE_68255620268124
// MI455X (gfx1250) — matrix operand order repaired and checked by reading, not compiled, not run on hardware
//
#include <hip/hip_runtime.h>

// ---------------------------------------------------------------------------
// Types
// ---------------------------------------------------------------------------
typedef __attribute__((ext_vector_type(16))) __bf16 v16bf;
typedef __attribute__((ext_vector_type(8)))  __bf16 v8bf;
typedef __attribute__((ext_vector_type(8)))  float  v8f;
typedef __attribute__((ext_vector_type(4)))  int    v4i;

union FragAB { v16bf v; v8bf h[2]; };

__device__ __forceinline__ float bf2f(__bf16 x) {
    unsigned short u; __builtin_memcpy(&u, &x, 2);
    unsigned int w = ((unsigned int)u) << 16;
    float f; __builtin_memcpy(&f, &w, 4);
    return f;
}
__device__ __forceinline__ float to_f32(float x) { return x; }
__device__ __forceinline__ float to_f32(__bf16 x) { return bf2f(x); }

__device__ __forceinline__ __bf16 f2bf(float f) {
    unsigned int u; __builtin_memcpy(&u, &f, 4);
    u += 0x7FFFu + ((u >> 16) & 1u);           // round-to-nearest-even
    unsigned short h = (unsigned short)(u >> 16);
    __bf16 r; __builtin_memcpy(&r, &h, 2);
    return r;
}

// ---------------------------------------------------------------------------
// CDNA5 async global->LDS path (ASYNCcnt-tracked), with sync fallback.
// Probe-derived prototype: (v4i addrspace(1)*, v4i addrspace(3)*, i32, i32)
// ---------------------------------------------------------------------------
#if defined(__gfx1250__) && __has_builtin(__builtin_amdgcn_global_load_async_to_lds_b128) && __has_builtin(__builtin_amdgcn_s_wait_asynccnt)
#define HAVE_ASYNC_LDS 1
#else
#define HAVE_ASYNC_LDS 0
#endif

typedef __attribute__((address_space(1))) v4i gv4i;
typedef __attribute__((address_space(3))) v4i lv4i;

__device__ __forceinline__ void cp16_async(const __bf16* g, __bf16* l) {
#if HAVE_ASYNC_LDS
    __builtin_amdgcn_global_load_async_to_lds_b128(
        (gv4i*)(__bf16*)g,   // cast away const, generic -> global AS
        (lv4i*)l,            // generic -> LDS AS
        0, 0);
#else
    *(v8bf*)l = *(const v8bf*)g;
#endif
}

// ---------------------------------------------------------------------------
// Problem constants
// ---------------------------------------------------------------------------
#define DIMD   1024
#define SEQ    4096
#define BATCH  4
#define NEXP   8
#define NSLOT  512
#define ES     4096   /* NEXP*NSLOT */
#define HID    4096

// ---------------------------------------------------------------------------
// RMSNorm (l2-normalize row, * sqrt(d) * gamma) -> bf16
// one block per row of `cols` (=1024), 256 threads x 4 elems
// ---------------------------------------------------------------------------
__global__ __launch_bounds__(256) void rmsnorm_bf16_k(
    const float* __restrict__ x, const float* __restrict__ gamma,
    __bf16* __restrict__ out, int cols)
{
    int row = blockIdx.x, tid = threadIdx.x;
    const float* p = x + (size_t)row * cols;
    float v[4]; float ss = 0.f;
#pragma unroll
    for (int j = 0; j < 4; j++) { v[j] = p[tid + j * 256]; ss += v[j] * v[j]; }
    __shared__ float red[256];
    red[tid] = ss; __syncthreads();
    for (int s = 128; s > 0; s >>= 1) { if (tid < s) red[tid] += red[tid + s]; __syncthreads(); }
    float norm = sqrtf(red[0]);
    float scale = sqrtf((float)cols) / fmaxf(norm, 1e-12f);
    __bf16* o = out + (size_t)row * cols;
#pragma unroll
    for (int j = 0; j < 4; j++) {
        int c = tid + j * 256;
        o[c] = f2bf(v[j] * scale * gamma[c]);
    }
}

// ---------------------------------------------------------------------------
// Tiled transpose: in [R,C] (f32 or bf16) -> out [C,R] bf16, batched over z
// ---------------------------------------------------------------------------
template<typename Tin>
__global__ __launch_bounds__(256) void transpose_to_bf16_k(
    const Tin* __restrict__ in, __bf16* __restrict__ out,
    int R, int C, long long strIn, long long strOut)
{
    __shared__ float t[32][33];
    int z = blockIdx.z;
    const Tin* iz = in + (size_t)z * strIn;
    __bf16*    oz = out + (size_t)z * strOut;
    int c0 = blockIdx.x * 32, r0 = blockIdx.y * 32;
    int c = threadIdx.x, ry = threadIdx.y;           // block (32,8)
#pragma unroll
    for (int rr = ry; rr < 32; rr += 8)
        t[rr][c] = to_f32(iz[(size_t)(r0 + rr) * C + c0 + c]);
    __syncthreads();
#pragma unroll
    for (int rr = ry; rr < 32; rr += 8)
        oz[(size_t)(c0 + rr) * R + r0 + c] = f2bf(t[c][rr]);
}

// ---------------------------------------------------------------------------
// Column softmax stats over n (axis=1): per (b, es) online max / 1/sum
// grid (ES/256, BATCH), 256 threads, thread owns one es column
// ---------------------------------------------------------------------------
__global__ __launch_bounds__(256) void col_softmax_stats_k(
    const float* __restrict__ L, float* __restrict__ cmax, float* __restrict__ csumr,
    int rows, int cols)
{
    int b = blockIdx.y;
    int es = blockIdx.x * 256 + threadIdx.x;
    const float* p = L + (size_t)b * rows * cols + es;
    float m = -3.0e38f, s = 0.f;
    for (int n = 0; n < rows; n++) {
        float v = p[(size_t)n * cols];
        float nm = fmaxf(m, v);
        s = s * __expf(m - nm) + __expf(v - nm);
        m = nm;
    }
    cmax[(size_t)b * cols + es] = m;
    csumr[(size_t)b * cols + es] = 1.0f / s;
}

// ---------------------------------------------------------------------------
// dispatch = exp(L - colmax)*colsumr, transposed to [b, es, n] bf16
// grid (ES/32, SEQ/32, BATCH), block (32,8)
// ---------------------------------------------------------------------------
__global__ __launch_bounds__(256) void dispatch_transpose_k(
    const float* __restrict__ L, const float* __restrict__ cmax,
    const float* __restrict__ csumr, __bf16* __restrict__ DT, int Nn, int Nes)
{
    __shared__ float t[32][33];
    int b = blockIdx.z;
    int es0 = blockIdx.x * 32, n0 = blockIdx.y * 32;
    int c = threadIdx.x, ry = threadIdx.y;
    float cm = cmax[(size_t)b * Nes + es0 + c];
    float cr = csumr[(size_t)b * Nes + es0 + c];
#pragma unroll
    for (int rr = ry; rr < 32; rr += 8) {
        float v = L[((size_t)b * Nn + n0 + rr) * Nes + es0 + c];
        t[rr][c] = __expf(v - cm) * cr;
    }
    __syncthreads();
#pragma unroll
    for (int rr = ry; rr < 32; rr += 8)
        DT[((size_t)b * Nes + es0 + rr) * Nn + n0 + c] = f2bf(t[c][rr]);
}

// ---------------------------------------------------------------------------
// Row softmax over last dim -> bf16 combine weights. Block per row, 4096 cols
// ---------------------------------------------------------------------------
__global__ __launch_bounds__(256) void row_softmax_bf16_k(
    const float* __restrict__ L, __bf16* __restrict__ out, int cols)
{
    int row = blockIdx.x, tid = threadIdx.x;
    const float* p = L + (size_t)row * cols;
    float v[16];
    float m = -3.0e38f;
#pragma unroll
    for (int j = 0; j < 16; j++) { v[j] = p[tid + j * 256]; m = fmaxf(m, v[j]); }
    __shared__ float red[256];
    red[tid] = m; __syncthreads();
    for (int s = 128; s > 0; s >>= 1) { if (tid < s) red[tid] = fmaxf(red[tid], red[tid + s]); __syncthreads(); }
    m = red[0]; __syncthreads();
    float s = 0.f;
#pragma unroll
    for (int j = 0; j < 16; j++) { v[j] = __expf(v[j] - m); s += v[j]; }
    red[tid] = s; __syncthreads();
    for (int t2 = 128; t2 > 0; t2 >>= 1) { if (tid < t2) red[tid] += red[tid + t2]; __syncthreads(); }
    float rs = 1.0f / red[0];
    __bf16* o = out + (size_t)row * cols;
#pragma unroll
    for (int j = 0; j < 16; j++) o[tid + j * 256] = f2bf(v[j] * rs);
}

// ---------------------------------------------------------------------------
// BF16 TN GEMM: C[M,N] = A[M,K] * Bt[N,K]^T  (both K-contiguous), WMMA bf16
// Block tile 128x128x32, 256 threads = 8 wave32 (2 along M x 4 along N),
// each wave: 4x2 tiles of 16x16 -> 8 v_wmma_f32_16x16x32_bf16 per K-step.
// Double-buffered LDS; stages prefetched with GLOBAL_LOAD_ASYNC_TO_LDS_B128
// (ASYNCcnt) so the next tile streams in while WMMAs run.
// EPI: 0 = store f32; 1 = store bf16; 2 = +bias, bf16; 3 = +bias, GELU, bf16
// ---------------------------------------------------------------------------
template<int EPI>
__global__ __launch_bounds__(256) void gemm_tn_k(
    const __bf16* __restrict__ A, const __bf16* __restrict__ Bt,
    void* __restrict__ Cv, const float* __restrict__ bias,
    int M, int N, int K,
    long long strA, long long strB, long long strC,
    int bmod, long long strBias)
{
    constexpr int BM = 128, BN = 128, BK = 32, LDT = BK + 8;  // 40 bf16 = 80B rows (16B aligned)
    __shared__ __bf16 sA[2][BM * LDT];
    __shared__ __bf16 sB[2][BN * LDT];

    int z = blockIdx.z;
    const __bf16* Az = A + (size_t)z * strA;
    const __bf16* Bz = Bt + (size_t)(z % bmod) * strB;
    const float* biasz = (EPI >= 2) ? (bias + (size_t)(z % bmod) * strBias) : nullptr;

    int n0 = blockIdx.x * BN, m0 = blockIdx.y * BM;
    int tid = threadIdx.x;
    int lane = tid & 31, wave = tid >> 5;
    int waveM = wave & 1, waveN = wave >> 1;   // 2 x 4 wave grid
    int l16 = lane & 15, hi = lane >> 4;

    v8f acc[4][2] = {};

    // staging map: thread -> (row krow & krow+64, 16B chunk kch); straight-line,
    // no data-dependent trip counts -> no exec divergence
    int krow = tid >> 2;            // 0..63
    int kch = (tid & 3) << 3;       // 0,8,16,24 (bf16 elems)
    const __bf16* gA0 = Az + (size_t)(m0 + krow) * K + kch;
    const __bf16* gA1 = gA0 + (size_t)64 * K;
    const __bf16* gB0 = Bz + (size_t)(n0 + krow) * K + kch;
    const __bf16* gB1 = gB0 + (size_t)64 * K;
    int lo0 = krow * LDT + kch;
    int lo1 = (krow + 64) * LDT + kch;

    auto stage = [&](int buf, int kt) {
        cp16_async(gA0 + kt, &sA[buf][lo0]);
        cp16_async(gA1 + kt, &sA[buf][lo1]);
        cp16_async(gB0 + kt, &sB[buf][lo0]);
        cp16_async(gB1 + kt, &sB[buf][lo1]);
    };

    int nk = K / BK;
    stage(0, 0);

    for (int i = 0; i < nk; i++) {
        int cur = i & 1;
        bool pre = (i + 1 < nk);
        if (pre) stage(cur ^ 1, (i + 1) * BK);   // prefetch next stage into other buffer
#if HAVE_ASYNC_LDS
        // retire previous stage's 4 async ops (loads complete in order);
        // leave the just-issued 4 in flight
        if (pre) __builtin_amdgcn_s_wait_asynccnt(4);
        else     __builtin_amdgcn_s_wait_asynccnt(0);
#endif
        __syncthreads();

        // ---- fragments in ISA layout ----
        FragAB fa[4], fb[2];
#pragma unroll
        for (int mi = 0; mi < 4; mi++) {
            // A lane: row M = l16 ; K = {0..7,16..23} (lanes<16) / {8..15,24..31}
            const __bf16* pa = &sA[cur][(waveM * 64 + mi * 16 + l16) * LDT + hi * 8];
            fa[mi].h[0] = *(const v8bf*)pa;
            fa[mi].h[1] = *(const v8bf*)(pa + 16);
        }
#pragma unroll
        for (int ni = 0; ni < 2; ni++) {
            // B lane: col N = l16 ; K = {0..7,16..23} (lanes<16) / {8..15,24..31}
            const __bf16* pb = &sB[cur][(waveN * 32 + ni * 16 + l16) * LDT + hi * 8];
            fb[ni].h[0] = *(const v8bf*)pb;
            fb[ni].h[1] = *(const v8bf*)(pb + 16);
        }
#pragma unroll
        for (int mi = 0; mi < 4; mi++)
#pragma unroll
            for (int ni = 0; ni < 2; ni++)
                acc[mi][ni] = __builtin_amdgcn_wmma_f32_16x16x32_bf16(
                    false, fa[mi].v, false, fb[ni].v, (short)0, acc[mi][ni], false, false);
        __syncthreads();   // all reads of buf `cur` done before it is re-staged
    }

    // ---- epilogue: VGPR r, lanes<16 -> M=r, lanes>=16 -> M=r+8 ; N = l16 ----
#pragma unroll
    for (int mi = 0; mi < 4; mi++) {
        int mbase = m0 + waveM * 64 + mi * 16 + hi * 8;
#pragma unroll
        for (int ni = 0; ni < 2; ni++) {
            int n = n0 + waveN * 32 + ni * 16 + l16;
            float bv = 0.f;
            if (EPI >= 2) bv = biasz[n];
#pragma unroll
            for (int r = 0; r < 8; r++) {
                float v = acc[mi][ni][r];
                if (EPI >= 2) v += bv;
                if (EPI == 3) v = 0.5f * v * (1.0f + erff(v * 0.70710678118654752f));
                size_t off = (size_t)z * strC + (size_t)(mbase + r) * N + n;
                if (EPI == 0) ((float*)Cv)[off] = v;
                else          ((__bf16*)Cv)[off] = f2bf(v);
            }
        }
    }
}

// ---------------------------------------------------------------------------
// Host-side orchestration
// ---------------------------------------------------------------------------
extern "C" void kernel_launch(void* const* d_in, const int* in_sizes, int n_in,
                              void* d_out, int out_size, void* d_ws, size_t ws_size,
                              hipStream_t stream) {
    (void)in_sizes; (void)n_in; (void)out_size; (void)ws_size;
    const float* x        = (const float*)d_in[0];   // [4,4096,1024]
    const float* gamma    = (const float*)d_in[1];   // [1024]
    const float* sgamma   = (const float*)d_in[2];   // [1024]
    const float* sembeds  = (const float*)d_in[3];   // [8,512,1024]
    const float* w1       = (const float*)d_in[4];   // [8,1024,4096]
    const float* b1       = (const float*)d_in[5];   // [8,4096]
    const float* w2       = (const float*)d_in[6];   // [8,4096,1024]
    const float* b2       = (const float*)d_in[7];   // [8,1024]
    float* out            = (float*)d_out;           // [4,4096,1024]

    // workspace carve-up
    char* w = (char*)d_ws;
    auto alloc = [&](size_t bytes) -> void* {
        void* p = (void*)w;
        w += (bytes + 255) & ~(size_t)255;
        return p;
    };
    typedef long long ll;
    __bf16* xn      = (__bf16*)alloc((size_t)BATCH * SEQ * DIMD * 2);    // [b,n,d]
    __bf16* se      = (__bf16*)alloc((size_t)ES * DIMD * 2);             // [es,d]
    __bf16* xnT     = (__bf16*)alloc((size_t)BATCH * DIMD * SEQ * 2);    // [b,d,n]
    __bf16* w1t     = (__bf16*)alloc((size_t)NEXP * HID * DIMD * 2);     // [e,h,d]
    __bf16* w2t     = (__bf16*)alloc((size_t)NEXP * DIMD * HID * 2);     // [e,d,h]
    float*  L       = (float*) alloc((size_t)BATCH * SEQ * ES * 4);      // logits [b,n,es]
    float*  cmax    = (float*) alloc((size_t)BATCH * ES * 4);
    float*  csumr   = (float*) alloc((size_t)BATCH * ES * 4);
    __bf16* DT      = (__bf16*)alloc((size_t)BATCH * ES * SEQ * 2);      // dispatch^T, later combine
    __bf16* slots   = (__bf16*)alloc((size_t)BATCH * ES * DIMD * 2);     // [b,es,d]
    __bf16* hbuf    = (__bf16*)alloc((size_t)BATCH * ES * HID * 2);      // [b,e,s,h]
    __bf16* oslots  = (__bf16*)alloc((size_t)BATCH * ES * DIMD * 2);     // [b,es,d]
    __bf16* oslotsT = (__bf16*)alloc((size_t)BATCH * DIMD * ES * 2);     // [b,d,es]

    // 1) RMSNorm x and slot_embeds -> bf16
    rmsnorm_bf16_k<<<BATCH * SEQ, 256, 0, stream>>>(x, gamma, xn, DIMD);
    rmsnorm_bf16_k<<<ES, 256, 0, stream>>>(sembeds, sgamma, se, DIMD);

    // 2) transposes: xn^T per batch; w1 -> [e,h,d]; w2 -> [e,d,h]
    transpose_to_bf16_k<__bf16><<<dim3(DIMD/32, SEQ/32, BATCH), dim3(32,8), 0, stream>>>(
        xn, xnT, SEQ, DIMD, (ll)SEQ*DIMD, (ll)DIMD*SEQ);
    transpose_to_bf16_k<float><<<dim3(HID/32, DIMD/32, NEXP), dim3(32,8), 0, stream>>>(
        w1, w1t, DIMD, HID, (ll)DIMD*HID, (ll)HID*DIMD);
    transpose_to_bf16_k<float><<<dim3(DIMD/32, HID/32, NEXP), dim3(32,8), 0, stream>>>(
        w2, w2t, HID, DIMD, (ll)HID*DIMD, (ll)DIMD*HID);

    // 3) logits[b] = xn[b] (4096x1024) * se^T (1024x4096) -> f32
    gemm_tn_k<0><<<dim3(ES/128, SEQ/128, BATCH), 256, 0, stream>>>(
        xn, se, L, nullptr, SEQ, ES, DIMD,
        (ll)SEQ*DIMD, 0, (ll)SEQ*ES, 1, 0);

    // 4) dispatch softmax over n: column stats then exp-normalize-transpose
    col_softmax_stats_k<<<dim3(ES/256, BATCH), 256, 0, stream>>>(L, cmax, csumr, SEQ, ES);
    dispatch_transpose_k<<<dim3(ES/32, SEQ/32, BATCH), dim3(32,8), 0, stream>>>(
        L, cmax, csumr, DT, SEQ, ES);

    // 5) slots[b] (4096es x 1024d) = DT[b] (4096x4096) * xnT[b]^T
    gemm_tn_k<1><<<dim3(DIMD/128, ES/128, BATCH), 256, 0, stream>>>(
        DT, xnT, slots, nullptr, ES, DIMD, SEQ,
        (ll)ES*SEQ, (ll)DIMD*SEQ, (ll)ES*DIMD, BATCH, 0);

    // 6) FFN1 per (b,e): h = GELU(slots * w1 + b1), z = b*8+e, expert = z%8
    gemm_tn_k<3><<<dim3(HID/128, NSLOT/128, BATCH*NEXP), 256, 0, stream>>>(
        slots, w1t, hbuf, b1, NSLOT, HID, DIMD,
        (ll)NSLOT*DIMD, (ll)HID*DIMD, (ll)NSLOT*HID, NEXP, (ll)HID);

    // 7) FFN2 per (b,e): out_slots = h * w2 + b2
    gemm_tn_k<2><<<dim3(DIMD/128, NSLOT/128, BATCH*NEXP), 256, 0, stream>>>(
        hbuf, w2t, oslots, b2, NSLOT, DIMD, HID,
        (ll)NSLOT*HID, (ll)DIMD*HID, (ll)NSLOT*DIMD, NEXP, (ll)DIMD);

    // 8) combine softmax over es (rows of L) -> bf16, reusing DT buffer
    row_softmax_bf16_k<<<BATCH * SEQ, 256, 0, stream>>>(L, DT, ES);

    // 9) out_slots^T per batch
    transpose_to_bf16_k<__bf16><<<dim3(DIMD/32, ES/32, BATCH), dim3(32,8), 0, stream>>>(
        oslots, oslotsT, ES, DIMD, (ll)ES*DIMD, (ll)DIMD*ES);

    // 10) out[b] (4096n x 1024d) = combine[b] (4096x4096) * oslotsT[b]^T  -> f32
    gemm_tn_k<0><<<dim3(DIMD/128, SEQ/128, BATCH), 256, 0, stream>>>(
        DT, oslotsT, out, nullptr, SEQ, DIMD, ES,
        (ll)SEQ*ES, (ll)DIMD*ES, (ll)SEQ*DIMD, BATCH, 0);
}
